// PackedTernaryWeight_19653770346751
// MI455X (gfx1250) — matrix operand order repaired and checked by reading, not compiled, not run on hardware
//
#include <hip/hip_runtime.h>
#include <cstdint>

typedef __attribute__((ext_vector_type(16))) __bf16 v16bf;
typedef __attribute__((ext_vector_type(8)))  float  v8f;

#define IN_F      4096
#define OUT_F     4096
#define PACKED_K  1024
#define BM        128
#define BN        128
#define BK        64
#define LSTR      72   // ushorts per LDS row = 144 bytes (16B aligned), pads banks

union Frag   { v16bf v; uint4 q[2]; };
union Pack32 { __bf16 h[32]; uint16_t u[32]; uint4 q[4]; };

// sched_group_barrier masks
#define SG_WMMA    0x008
#define SG_DSREAD  0x100

__global__ __launch_bounds__(256)
void ternary_linear_wmma(const float* __restrict__ x,
                         const uint8_t* __restrict__ pw,
                         const float* __restrict__ scale,
                         const float* __restrict__ bias,
                         float* __restrict__ out)
{
    __shared__ uint16_t As[BM * LSTR];   // bf16 bits, x tile (M x K)
    __shared__ uint16_t Bs[BN * LSTR];   // bf16 bits, W tile (N x K) == column-major B

    const int tid  = threadIdx.x;
    const int lane = tid & 31;
    const int wv   = tid >> 5;
    const int lhi  = lane >> 4;          // 0 | 1
    const int llo  = lane & 15;

    const int m0 = blockIdx.y * BM;
    const int n0 = blockIdx.x * BN;

    // ---- cooperative fill: 2 threads per row, each covers 32 K ----
    const int arow = tid >> 1;
    const int acol = (tid & 1) << 5;                 // 0 or 32 (floats)
    const float*   gA = x  + (size_t)(m0 + arow) * IN_F + acol;
    const int bby  = (tid & 1) << 3;                 // 0 or 8 (bytes = 32 K)
    const uint8_t* gB = pw + (size_t)(n0 + arow) * PACKED_K + bby;

    // ---- wave tile: 2x4 wave grid, each wave 64(M) x 32(N) ----
    const int wm = (wv >> 2) * 64;
    const int wn = (wv & 3) * 32;

    v8f acc[4][2] = {};

    // register staging for the software pipeline
    float4   sa[8];
    uint32_t sb[2];

    // prologue: stage K-tile 0
#pragma unroll
    for (int i = 0; i < 8; ++i) sa[i] = ((const float4*)gA)[i];
    sb[0] = ((const uint32_t*)gB)[0];
    sb[1] = ((const uint32_t*)gB)[1];

    for (int k0 = 0; k0 < IN_F; k0 += BK) {
        __syncthreads();                 // previous iteration's consumers done

        // ---------- write staged tile to LDS ----------
        {   // A: fp32 -> bf16 (v_cvt_pk_bf16_f32)
            Pack32 ha;
            const float* fs = (const float*)sa;
#pragma unroll
            for (int j = 0; j < 32; ++j) ha.h[j] = (__bf16)fs[j];
            uint4* dst = (uint4*)&As[arow * LSTR + acol];
#pragma unroll
            for (int i = 0; i < 4; ++i) dst[i] = ha.q[i];
        }
        {   // B: 2-bit ternary -> bf16 bits {0xBF80, 0, 0x3F80}
            Pack32 hb;
#pragma unroll
            for (int half = 0; half < 2; ++half) {
                const uint32_t w = sb[half];
#pragma unroll
                for (int j = 0; j < 16; ++j) {
                    const uint32_t f = (w >> (2 * j)) & 3u;
                    hb.u[half * 16 + j] =
                        (f == 0u) ? 0xBF80u : ((f == 2u) ? 0x3F80u : 0u);
                }
            }
            uint4* dst = (uint4*)&Bs[arow * LSTR + (bby << 2)];
#pragma unroll
            for (int i = 0; i < 4; ++i) dst[i] = hb.q[i];
        }

        __syncthreads();                 // tile visible to all waves

        // ---------- stage next K-tile (latency hides behind the WMMAs) ----------
        if (k0 + BK < IN_F) {
            const float* pA = gA + (k0 + BK);
#pragma unroll
            for (int i = 0; i < 8; ++i) sa[i] = ((const float4*)pA)[i];
            const uint8_t* pB = gB + ((k0 + BK) >> 2);
            sb[0] = ((const uint32_t*)pB)[0];
            sb[1] = ((const uint32_t*)pB)[1];
            if (k0 + 2 * BK < IN_F)
                __builtin_prefetch((const void*)(gA + k0 + 2 * BK), 0, 0);
        }

        // ---------- 2 K-halves x 8 WMMAs each ----------
#pragma unroll
        for (int h = 0; h < 2; ++h) {
            const int kb = h * 32;       // ushort offset within the row

            Frag fa0, fa1, fa2, fa3, fb0, fb1;
            {
                // A 16x32: lanes 0-15 K=0..7 & 16..23, lanes 16-31 K=8..15 & 24..31
                const uint16_t* r0 = &As[(wm +  0 + llo) * LSTR + kb];
                const uint16_t* r1 = &As[(wm + 16 + llo) * LSTR + kb];
                const uint16_t* r2 = &As[(wm + 32 + llo) * LSTR + kb];
                const uint16_t* r3 = &As[(wm + 48 + llo) * LSTR + kb];
                fa0.q[0] = *(const uint4*)(r0 + lhi * 8);
                fa0.q[1] = *(const uint4*)(r0 + 16 + lhi * 8);
                fa1.q[0] = *(const uint4*)(r1 + lhi * 8);
                fa1.q[1] = *(const uint4*)(r1 + 16 + lhi * 8);
                fa2.q[0] = *(const uint4*)(r2 + lhi * 8);
                fa2.q[1] = *(const uint4*)(r2 + 16 + lhi * 8);
                fa3.q[0] = *(const uint4*)(r3 + lhi * 8);
                fa3.q[1] = *(const uint4*)(r3 + 16 + lhi * 8);
                // B 32x16: lanes 0-15 K=0..7 & 16..23, lanes 16-31 K=8..15 & 24..31 of column N=llo
                const uint16_t* s0 = &Bs[(wn +  0 + llo) * LSTR + kb];
                const uint16_t* s1 = &Bs[(wn + 16 + llo) * LSTR + kb];
                fb0.q[0] = *(const uint4*)(s0 + lhi * 8);
                fb0.q[1] = *(const uint4*)(s0 + 16 + lhi * 8);
                fb1.q[0] = *(const uint4*)(s1 + lhi * 8);
                fb1.q[1] = *(const uint4*)(s1 + 16 + lhi * 8);
            }

            acc[0][0] = __builtin_amdgcn_wmma_f32_16x16x32_bf16(false, fa0.v, false, fb0.v, (short)0, acc[0][0], false, false);
            acc[0][1] = __builtin_amdgcn_wmma_f32_16x16x32_bf16(false, fa0.v, false, fb1.v, (short)0, acc[0][1], false, false);
            acc[1][0] = __builtin_amdgcn_wmma_f32_16x16x32_bf16(false, fa1.v, false, fb0.v, (short)0, acc[1][0], false, false);
            acc[1][1] = __builtin_amdgcn_wmma_f32_16x16x32_bf16(false, fa1.v, false, fb1.v, (short)0, acc[1][1], false, false);
            acc[2][0] = __builtin_amdgcn_wmma_f32_16x16x32_bf16(false, fa2.v, false, fb0.v, (short)0, acc[2][0], false, false);
            acc[2][1] = __builtin_amdgcn_wmma_f32_16x16x32_bf16(false, fa2.v, false, fb1.v, (short)0, acc[2][1], false, false);
            acc[3][0] = __builtin_amdgcn_wmma_f32_16x16x32_bf16(false, fa3.v, false, fb0.v, (short)0, acc[3][0], false, false);
            acc[3][1] = __builtin_amdgcn_wmma_f32_16x16x32_bf16(false, fa3.v, false, fb1.v, (short)0, acc[3][1], false, false);

            // ask the scheduler for: all 12 DS reads first, then the 8 WMMAs
            __builtin_amdgcn_sched_group_barrier(SG_DSREAD, 12, 0);
            __builtin_amdgcn_sched_group_barrier(SG_WMMA,    8, 0);
        }
    }

    // ---------- epilogue: fuse per-channel scale + bias ----------
#pragma unroll
    for (int nt = 0; nt < 2; ++nt) {
        const int gn = n0 + wn + nt * 16 + llo;
        const float s  = scale[gn];
        const float bv = bias[gn];
#pragma unroll
        for (int mt = 0; mt < 4; ++mt) {
#pragma unroll
            for (int r = 0; r < 8; ++r) {
                const int gm = m0 + wm + mt * 16 + lhi * 8 + r;
                out[(size_t)gm * OUT_F + gn] = acc[mt][nt][r] * s + bv;
            }
        }
    }
}

extern "C" void kernel_launch(void* const* d_in, const int* in_sizes, int n_in,
                              void* d_out, int out_size, void* d_ws, size_t ws_size,
                              hipStream_t stream) {
    const float*   x     = (const float*)d_in[0];
    const uint8_t* pw    = (const uint8_t*)d_in[1];
    const float*   scale = (const float*)d_in[2];
    const float*   bias  = (const float*)d_in[3];
    float*         out   = (float*)d_out;

    const int M = in_sizes[0] / IN_F;               // 8192
    dim3 grid(OUT_F / BN, M / BM);                  // (32, 64)
    ternary_linear_wmma<<<grid, 256, 0, stream>>>(x, pw, scale, bias, out);
}
